// STDPPlasticity_17274358464692
// MI455X (gfx1250) — matrix operand order repaired and checked by reading, not compiled, not run on hardware
//
#include <hip/hip_runtime.h>

typedef __attribute__((ext_vector_type(16))) _Float16 v16h;
typedef __attribute__((ext_vector_type(2)))  _Float16 v2h;
typedef __attribute__((ext_vector_type(8)))  float    v8f;
typedef __attribute__((ext_vector_type(4)))  float    v4f;

#define OUT_DIM 8192
#define IN_DIM  8192
#define BATCH   32
#define LR      0.01f

// ---------------------------------------------------------------------------
// Prep: f32 -> f16 transposed copies of the activations, laid out [row][batch]
// so both WMMA operand fragments load as contiguous 32-bit f16 pairs.
// Also zeroes the global NaN flag (ws is poisoned by the harness).
// ---------------------------------------------------------------------------
__global__ void stdp_prep(const float* __restrict__ pre,
                          const float* __restrict__ post,
                          _Float16* __restrict__ preT,
                          _Float16* __restrict__ postT,
                          unsigned int* __restrict__ flag) {
  unsigned tid = blockIdx.x * blockDim.x + threadIdx.x;   // 0 .. 8192*32-1
  if (tid == 0) *flag = 0u;
  unsigned row = tid >> 5;
  unsigned b   = tid & 31;
  preT[tid]  = (_Float16)pre [b * (unsigned)IN_DIM  + row];
  postT[tid] = (_Float16)post[b * (unsigned)OUT_DIM + row];
}

// ---------------------------------------------------------------------------
// Main: each wave computes a 16(O) x 64(I) strip = 4 WMMA tiles.
//   delta_tile = A(post^T 16x32 f16) x B(pre 32x16 f16)   [one WMMA per tile]
//   out = clamp(w + (LR/B) * delta, 0, 1), non-temporal streaming of weights.
// Block = 256 threads = 8 waves stacked along O -> block tile 128(O) x 64(I).
// ---------------------------------------------------------------------------
__global__ __launch_bounds__(256) void stdp_hebbian(
    const float* __restrict__ W,
    const _Float16* __restrict__ preT,
    const _Float16* __restrict__ postT,
    float* __restrict__ Wout,
    unsigned int* __restrict__ flag) {
  const int lane = threadIdx.x & 31;
  const int wv   = threadIdx.x >> 5;
  const int n    = lane & 15;   // N column (B/C/D) == M row (A) for this lane
  const int hi   = lane >> 4;   // half-wave select
  const int o0   = blockIdx.y * 128 + wv * 16;
  const int i0   = blockIdx.x * 64;

  // --- A fragment: 16x32 f16, rows of post^T. ISA layout (05_wmma §7.12.2):
  //     VGPR v, half hi -> K = 2*(v&3) + 16*(v>>2) + 8*hi (packed pair).
  v16h Af;
  {
    const _Float16* arow = postT + (size_t)(o0 + n) * BATCH;
    #pragma unroll
    for (int v = 0; v < 8; ++v) {
      const int k0 = 2 * (v & 3) + ((v >> 2) << 4) + (hi << 3);
      v2h p = *(const v2h*)(arow + k0);
      Af[2 * v]     = p.x;
      Af[2 * v + 1] = p.y;
    }
  }

  const float S = LR / (float)BATCH;
  bool sawNan = false;

  #pragma unroll
  for (int t = 0; t < 4; ++t) {
    const int ic = i0 + t * 16 + n;

    // --- B fragment: 32x16 f16. VGPR v, half hi -> K = 2*(v&3) + 16*(v>>2) + 8*hi.
    v16h Bf;
    const _Float16* brow = preT + (size_t)ic * BATCH;
    #pragma unroll
    for (int v = 0; v < 8; ++v) {
      const int k0 = 2 * (v & 3) + ((v >> 2) << 4) + (hi << 3);
      v2h p = *(const v2h*)(brow + k0);
      Bf[2 * v]     = p.x;
      Bf[2 * v + 1] = p.y;
    }

    // Entire K=32 reduction in one matrix op.
    v8f c = {};
    c = __builtin_amdgcn_wmma_f32_16x16x32_f16(
        /*neg_a=*/false, Af, /*neg_b=*/false, Bf,
        /*c_mod=*/(short)0, c, /*reuse_a=*/false, /*reuse_b=*/false);

    // C/D layout: VGPR v, lane -> row o0 + 8*hi + v, col ic. Stream weights NT.
    #pragma unroll
    for (int v = 0; v < 8; ++v) {
      const size_t idx = (size_t)(o0 + (hi << 3) + v) * IN_DIM + ic;
      const float w = __builtin_nontemporal_load(W + idx);
      sawNan |= (w != w);
      float r = fmaf(S, c[v], w);
      r = fminf(fmaxf(r, 0.0f), 1.0f);
      __builtin_nontemporal_store(r, Wout + idx);
    }
  }

  if (sawNan) atomicOr(flag, 1u);   // only executes if a NaN actually exists
}

// ---------------------------------------------------------------------------
// Fixup: reference semantics say ANY NaN in weights -> return weights
// unchanged. Flag is final once this launch runs (stream-ordered). Costs only
// launch overhead when no NaN is present.
// ---------------------------------------------------------------------------
__global__ void stdp_fixup(const float* __restrict__ W,
                           float* __restrict__ Wout,
                           const unsigned int* __restrict__ flag) {
  if (*flag == 0u) return;
  const size_t idx = ((size_t)blockIdx.x * blockDim.x + threadIdx.x) * 4;
  *(v4f*)(Wout + idx) = *(const v4f*)(W + idx);
}

extern "C" void kernel_launch(void* const* d_in, const int* in_sizes, int n_in,
                              void* d_out, int out_size, void* d_ws, size_t ws_size,
                              hipStream_t stream) {
  const float* W    = (const float*)d_in[0];   // weights      [8192, 8192]
  const float* pre  = (const float*)d_in[1];   // pre_activity [32, 8192]
  const float* post = (const float*)d_in[2];   // post_activity[32, 8192]
  float* Wout = (float*)d_out;

  const size_t halfBytes = (size_t)OUT_DIM * BATCH * sizeof(_Float16); // 512 KB
  _Float16* postT = (_Float16*)d_ws;
  _Float16* preT  = (_Float16*)((char*)d_ws + halfBytes);
  unsigned int* flag = (unsigned int*)((char*)d_ws + 2 * halfBytes);

  stdp_prep<<<(OUT_DIM * BATCH) / 256, 256, 0, stream>>>(pre, post, preT, postT, flag);

  dim3 grid(IN_DIM / 64, OUT_DIM / 128);
  stdp_hebbian<<<grid, dim3(256), 0, stream>>>(W, preT, postT, Wout, flag);

  stdp_fixup<<<((size_t)OUT_DIM * IN_DIM) / (256 * 4), 256, 0, stream>>>(W, Wout, flag);
}
